// WeightedMemory_1245540516317
// MI455X (gfx1250) — matrix operand order repaired and checked by reading, not compiled, not run on hardware
//
#include <hip/hip_runtime.h>

// out = (inputs / ||inputs||_row) @ features^T / TEMP ; TEMP = 0.05
#define TEMP_INV 20.0f
#define MBLK 4   // M-tiles (of 16 rows) per wave: 64x32 output tile per wave

typedef __bf16 v16bf __attribute__((ext_vector_type(16)));
typedef __bf16 v8bf  __attribute__((ext_vector_type(8)));
typedef float  v8f   __attribute__((ext_vector_type(8)));
typedef float  v4f   __attribute__((ext_vector_type(4)));

__device__ __forceinline__ v8f wmma_bf16(v16bf a, v16bf b, v8f c) {
  // 8 args: (neg_a, A, neg_b, B, c_mod, C, reuse_a, reuse_b)
  return __builtin_amdgcn_wmma_f32_16x16x32_bf16(false, a, false, b, (short)0, c,
                                                 false, false);
}

// Split 16 f32 into bf16 hi + bf16 residual (3-product split recovers ~f32).
__device__ __forceinline__ void split16(const float* __restrict__ v, v16bf& h,
                                        v16bf& l) {
#pragma unroll
  for (int i = 0; i < 16; ++i) {
    float f   = v[i];
    __bf16 hi = (__bf16)f;
    h[i]      = hi;
    l[i]      = (__bf16)(f - (float)hi);
  }
}

// ---------------- Kernel 1: row-normalize x, emit bf16 hi/lo planes ----------
__global__ void __launch_bounds__(256)
normalize_split_kernel(const float* __restrict__ x, __bf16* __restrict__ Xh,
                       __bf16* __restrict__ Xl, int B, int D) {
  const int wave = (int)((blockIdx.x * blockDim.x + threadIdx.x) >> 5);
  const int lane = threadIdx.x & 31;
  if (wave >= B) return;

  const int per    = D >> 5;  // D=256 -> 8 elements per lane
  const float* row = x + (size_t)wave * D;
  float vals[8];
  float s = 0.f;
  for (int i = 0; i < per; ++i) {
    float v = row[lane * per + i];
    vals[i] = v;
    s += v * v;
  }
  for (int off = 16; off > 0; off >>= 1) s += __shfl_xor(s, off, 32);
  const float inv = 1.0f / sqrtf(s);

  for (int i = 0; i < per; ++i) {
    float f    = vals[i] * inv;
    __bf16 hi  = (__bf16)f;
    size_t idx = (size_t)wave * D + lane * per + i;
    Xh[idx]    = hi;
    Xl[idx]    = (__bf16)(f - (float)hi);
  }
}

// ------------- Kernel 2: elementwise split of F into bf16 hi/lo planes -------
__global__ void __launch_bounds__(256)
convert_split_kernel(const float* __restrict__ F, __bf16* __restrict__ Fh,
                     __bf16* __restrict__ Fl, size_t total) {
  size_t i = ((size_t)blockIdx.x * blockDim.x + threadIdx.x) * 8;
  if (i >= total) return;
  v4f a = *(const v4f*)(F + i);
  v4f b = *(const v4f*)(F + i + 4);
  v8bf h, l;
#pragma unroll
  for (int j = 0; j < 4; ++j) {
    __bf16 hi = (__bf16)a[j];
    h[j]      = hi;
    l[j]      = (__bf16)(a[j] - (float)hi);
    __bf16 hj = (__bf16)b[j];
    h[4 + j]  = hj;
    l[4 + j]  = (__bf16)(b[j] - (float)hj);
  }
  *(v8bf*)(Fh + i) = h;
  *(v8bf*)(Fl + i) = l;
}

// --------- Kernel 3a: WMMA GEMM from precomputed bf16 planes (no VALU cvt) ---
// Block = 8 waves; wave tile = 64 rows (4 M-tiles) x 32 cols (2 subtiles).
// A (16x32 bf16): lanes 0-15 row M=l16; K = half*16..+16 (contiguous, matches B).
// B (32x16 bf16): lane = col N=l16, K = half*16..+16 -> contiguous row of F.
__global__ void __launch_bounds__(256)
gemm_planes_kernel(const __bf16* __restrict__ Xh, const __bf16* __restrict__ Xl,
                   const __bf16* __restrict__ Fh, const __bf16* __restrict__ Fl,
                   float* __restrict__ out, int N, int D) {
  const int lane = threadIdx.x & 31;
  const int wave = threadIdx.x >> 5;
  const int half = lane >> 4;
  const int l16  = lane & 15;
  const int m0   = blockIdx.y * (16 * MBLK);
  const int n0   = blockIdx.x * 256 + wave * 32;
  if (n0 >= N) return;  // N % 32 == 0 -> wave-uniform; EXEC stays all-1s

  v8f acc[MBLK][2];
#pragma unroll
  for (int m = 0; m < MBLK; ++m) {
    acc[m][0] = (v8f){};
    acc[m][1] = (v8f){};
  }

  const __bf16* bh0 = Fh + (size_t)(n0 + l16) * D + half * 16;
  const __bf16* bl0 = Fl + (size_t)(n0 + l16) * D + half * 16;
  const __bf16* bh1 = Fh + (size_t)(n0 + 16 + l16) * D + half * 16;
  const __bf16* bl1 = Fl + (size_t)(n0 + 16 + l16) * D + half * 16;

  for (int k = 0; k < D; k += 32) {
    v16bf B0h, B0l, B1h, B1l;
    ((v8bf*)&B0h)[0] = *(const v8bf*)(bh0 + k);
    ((v8bf*)&B0h)[1] = *(const v8bf*)(bh0 + k + 8);
    ((v8bf*)&B0l)[0] = *(const v8bf*)(bl0 + k);
    ((v8bf*)&B0l)[1] = *(const v8bf*)(bl0 + k + 8);
    ((v8bf*)&B1h)[0] = *(const v8bf*)(bh1 + k);
    ((v8bf*)&B1h)[1] = *(const v8bf*)(bh1 + k + 8);
    ((v8bf*)&B1l)[0] = *(const v8bf*)(bl1 + k);
    ((v8bf*)&B1l)[1] = *(const v8bf*)(bl1 + k + 8);

    if (k + 32 < D) {  // gfx1250 global_prefetch_b8
      __builtin_prefetch(bh0 + k + 32, 0, 1);
      __builtin_prefetch(bh1 + k + 32, 0, 1);
    }

#pragma unroll
    for (int m = 0; m < MBLK; ++m) {
      const size_t ar = (size_t)(m0 + 16 * m + l16) * D + k + half * 16;
      v16bf ah, al;
      ((v8bf*)&ah)[0] = *(const v8bf*)(Xh + ar);
      ((v8bf*)&ah)[1] = *(const v8bf*)(Xh + ar + 8);
      ((v8bf*)&al)[0] = *(const v8bf*)(Xl + ar);
      ((v8bf*)&al)[1] = *(const v8bf*)(Xl + ar + 8);

      acc[m][0] = wmma_bf16(ah, B0h, acc[m][0]);
      acc[m][0] = wmma_bf16(ah, B0l, acc[m][0]);
      acc[m][0] = wmma_bf16(al, B0h, acc[m][0]);
      acc[m][1] = wmma_bf16(ah, B1h, acc[m][1]);
      acc[m][1] = wmma_bf16(ah, B1l, acc[m][1]);
      acc[m][1] = wmma_bf16(al, B1h, acc[m][1]);
    }
  }

#pragma unroll
  for (int m = 0; m < MBLK; ++m) {
#pragma unroll
    for (int r = 0; r < 8; ++r) {
      size_t rowoff               = (size_t)(m0 + 16 * m + r + 8 * half) * N;
      out[rowoff + n0 + l16]      = acc[m][0][r] * TEMP_INV;
      out[rowoff + n0 + 16 + l16] = acc[m][1][r] * TEMP_INV;
    }
  }
}

// --------- Kernel 3b: fallback fused-conversion GEMM (small workspace) -------
__global__ void __launch_bounds__(256)
gemm_fused_kernel(const __bf16* __restrict__ Xh, const __bf16* __restrict__ Xl,
                  const float* __restrict__ F, float* __restrict__ out, int N,
                  int D) {
  const int lane = threadIdx.x & 31;
  const int wave = threadIdx.x >> 5;
  const int half = lane >> 4;
  const int l16  = lane & 15;
  const int m0   = blockIdx.y * (16 * MBLK);
  const int n0   = blockIdx.x * 256 + wave * 32;
  if (n0 >= N) return;

  v8f acc[MBLK][2];
#pragma unroll
  for (int m = 0; m < MBLK; ++m) {
    acc[m][0] = (v8f){};
    acc[m][1] = (v8f){};
  }

  const float* br0 = F + (size_t)(n0 + l16) * D + half * 16;
  const float* br1 = F + (size_t)(n0 + 16 + l16) * D + half * 16;

  for (int k = 0; k < D; k += 32) {
    v4f t4[4];
    float* t = (float*)t4;
    v16bf B0h, B0l, B1h, B1l;

    t4[0] = *(const v4f*)(br0 + k + 0);
    t4[1] = *(const v4f*)(br0 + k + 4);
    t4[2] = *(const v4f*)(br0 + k + 8);
    t4[3] = *(const v4f*)(br0 + k + 12);
    split16(t, B0h, B0l);

    t4[0] = *(const v4f*)(br1 + k + 0);
    t4[1] = *(const v4f*)(br1 + k + 4);
    t4[2] = *(const v4f*)(br1 + k + 8);
    t4[3] = *(const v4f*)(br1 + k + 12);
    split16(t, B1h, B1l);

#pragma unroll
    for (int m = 0; m < MBLK; ++m) {
      const size_t ar = (size_t)(m0 + 16 * m + l16) * D + k + half * 16;
      v16bf ah, al;
      ((v8bf*)&ah)[0] = *(const v8bf*)(Xh + ar);
      ((v8bf*)&ah)[1] = *(const v8bf*)(Xh + ar + 8);
      ((v8bf*)&al)[0] = *(const v8bf*)(Xl + ar);
      ((v8bf*)&al)[1] = *(const v8bf*)(Xl + ar + 8);

      acc[m][0] = wmma_bf16(ah, B0h, acc[m][0]);
      acc[m][0] = wmma_bf16(ah, B0l, acc[m][0]);
      acc[m][0] = wmma_bf16(al, B0h, acc[m][0]);
      acc[m][1] = wmma_bf16(ah, B1h, acc[m][1]);
      acc[m][1] = wmma_bf16(ah, B1l, acc[m][1]);
      acc[m][1] = wmma_bf16(al, B1h, acc[m][1]);
    }
  }

#pragma unroll
  for (int m = 0; m < MBLK; ++m) {
#pragma unroll
    for (int r = 0; r < 8; ++r) {
      size_t rowoff               = (size_t)(m0 + 16 * m + r + 8 * half) * N;
      out[rowoff + n0 + l16]      = acc[m][0][r] * TEMP_INV;
      out[rowoff + n0 + 16 + l16] = acc[m][1][r] * TEMP_INV;
    }
  }
}

extern "C" void kernel_launch(void* const* d_in, const int* in_sizes, int n_in,
                              void* d_out, int out_size, void* d_ws,
                              size_t ws_size, hipStream_t stream) {
  (void)n_in; (void)out_size;
  const float* x     = (const float*)d_in[0];
  // d_in[1] = targets: does not affect the returned tensor
  const float* feats = (const float*)d_in[2];

  const int B = in_sizes[1];      // 1024
  const int D = in_sizes[0] / B;  // 256
  const int N = in_sizes[2] / D;  // 100000

  const size_t planeF = (size_t)N * D;  // elements per F plane
  const size_t planeX = (size_t)B * D;  // elements per X plane
  const size_t needA  = (2 * planeF + 2 * planeX) * sizeof(__bf16);

  float* out = (float*)d_out;
  dim3 gemm_grid((unsigned)((N + 255) / 256), (unsigned)(B / (16 * MBLK)));
  dim3 nblk((unsigned)((B * 32 + 255) / 256));  // one wave per row

  if (ws_size >= needA) {
    // Workspace layout: Fh | Fl | Xh | Xl
    __bf16* Fh = reinterpret_cast<__bf16*>(d_ws);
    __bf16* Fl = Fh + planeF;
    __bf16* Xh = Fl + planeF;
    __bf16* Xl = Xh + planeX;

    normalize_split_kernel<<<nblk, 256, 0, stream>>>(x, Xh, Xl, B, D);
    convert_split_kernel<<<dim3((unsigned)((planeF / 8 + 255) / 256)), 256, 0,
                           stream>>>(feats, Fh, Fl, planeF);
    gemm_planes_kernel<<<gemm_grid, 256, 0, stream>>>(Xh, Xl, Fh, Fl, out, N, D);
  } else {
    // Small-workspace fallback: only X planes (1 MB), convert F on the fly.
    __bf16* Xh = reinterpret_cast<__bf16*>(d_ws);
    __bf16* Xl = Xh + planeX;

    normalize_split_kernel<<<nblk, 256, 0, stream>>>(x, Xh, Xl, B, D);
    gemm_fused_kernel<<<gemm_grid, 256, 0, stream>>>(Xh, Xl, feats, out, N, D);
  }
}
